// GcnDenseModel_55533927138057
// MI455X (gfx1250) — compile-verified, matrix operand order checked by reading, not run on hardware
//
#include <hip/hip_runtime.h>
#include <hip/hip_bf16.h>
#include <math.h>

// ---------------- problem constants (from reference) ----------------
#define B_   64
#define N_   256          // nodes per graph == GRU sequence length
#define FIN_ 256
#define OUT_ 256
#define DEP_ 64
#define HID_ 256
#define E_   (B_ * N_ * 16)   // 262144 edges
#define NN_  (B_ * N_)        // 16384 nodes

typedef __bf16 bf16;
typedef __attribute__((ext_vector_type(16))) __bf16 v16bf;
typedef __attribute__((ext_vector_type(8)))  float  v8f;

union FragBF { v16bf v; bf16 h[16]; uint4 q[2]; };

// A/B fragment for v_wmma_f32_16x16x32_bf16 from a row-major [rows,K] bf16
// matrix (ld = K). lane L -> row row0 + L%16 ; K elements kb..kb+7 and
// 16+kb..16+kb+7 with kb = (L/16)*8  (CDNA5 16-bit 16x32 layout).
static __device__ inline v16bf load_frag_bf16(const bf16* base, int ld, int row0, int k0) {
    const int lane = threadIdx.x & 31;
    const int r    = row0 + (lane & 15);
    const int kb   = (lane >> 4) << 3;
    FragBF f;
    f.q[0] = *(const uint4*)(base + (size_t)r * ld + k0 + kb);
    f.q[1] = *(const uint4*)(base + (size_t)r * ld + k0 + 16 + kb);
    return f.v;
}

static __device__ inline v8f wmma_bf16(v16bf a, v16bf b, v8f c) {
    return __builtin_amdgcn_wmma_f32_16x16x32_bf16(false, a, false, b, (short)0, c, false, false);
}

static __device__ inline float sigmoidf_(float x) { return 1.0f / (1.0f + __expf(-x)); }

// ---------------- small utility kernels ----------------
__global__ void k_fill_f32(float* p, float v, int n) {
    int i = blockIdx.x * blockDim.x + threadIdx.x;
    if (i < n) p[i] = v;
}

__global__ void k_convert_bf16(const float* __restrict__ in, bf16* __restrict__ out, int n) {
    int i = blockIdx.x * blockDim.x + threadIdx.x;
    if (i < n) out[i] = (bf16)in[i];
}

__global__ void k_edge_deg(const int* __restrict__ edge_index, float* __restrict__ deg) {
    int e = blockIdx.x * blockDim.x + threadIdx.x;
    if (e < E_) atomicAdd(&deg[edge_index[E_ + e]], 1.0f);  // col = second row
}

// Bones[n] = sum_k WB[n][k]  (self-loop edge_attr == ones)
__global__ void k_bones(const float* __restrict__ WB, float* __restrict__ Bones) {
    int n = threadIdx.x;
    float s = 0.f;
    #pragma unroll 8
    for (int k = 0; k < DEP_; ++k) s += WB[n * DEP_ + k];
    Bones[n] = s;
}

// ---------------- generic WMMA GEMM:  C[M,N] = A[M,K] @ W[N,K]^T (+bias) ----
// block = 128 threads (4 waves); block tile = 16 rows x 256 cols;
// grid = (M/16, N/256). Each wave owns a 16x64 strip (4 accum tiles).
__global__ __launch_bounds__(128)
void k_gemm_bf16(const bf16* __restrict__ A, const bf16* __restrict__ W,
                 const float* __restrict__ bias, float* __restrict__ C,
                 int M, int N, int K) {
    const int wave = threadIdx.x >> 5;
    const int lane = threadIdx.x & 31;
    const int m0 = blockIdx.x * 16;
    const int n0 = blockIdx.y * 256 + wave * 64;
    if (m0 >= M || n0 >= N) return;

    v8f acc[4] = {};
    for (int k0 = 0; k0 < K; k0 += 32) {
        v16bf a = load_frag_bf16(A, K, m0, k0);
        #pragma unroll
        for (int t = 0; t < 4; ++t) {
            v16bf b = load_frag_bf16(W, K, n0 + t * 16, k0);
            acc[t] = wmma_bf16(a, b, acc[t]);
        }
    }
    #pragma unroll
    for (int t = 0; t < 4; ++t) {
        #pragma unroll
        for (int i = 0; i < 8; ++i) {
            int row = m0 + i + ((lane >> 4) << 3);
            int col = n0 + t * 16 + (lane & 15);
            float v = acc[t][i];
            if (bias) v += bias[col];
            C[(size_t)row * N + col] = v;
        }
    }
}

// ---------------- fused edge-message kernel -------------------------------
// Block (128 thr, 4 waves) handles 16 edges. Wave w computes Be tile
// [16 edges x 64 cols] via WMMA (K=64), then msg = norm*tanh(Ax[row]*Be)
// scattered into agg[col] with f32 global atomics.
__global__ __launch_bounds__(128)
void k_edge_msg(const float* __restrict__ edge_attr, const int* __restrict__ edge_index,
                const float* __restrict__ deg, const float* __restrict__ Ax,
                const bf16* __restrict__ WBb, float* __restrict__ agg) {
    __shared__ int   srow[16], scol[16];
    __shared__ float snorm[16];
    const int e0  = blockIdx.x * 16;
    const int tid = threadIdx.x;
    if (tid < 16) {
        int e = e0 + tid;
        int r = edge_index[e];
        int c = edge_index[E_ + e];
        srow[tid] = r; scol[tid] = c;
        snorm[tid] = rsqrtf(deg[r]) * rsqrtf(deg[c]);
    }
    __syncthreads();

    const int wave = tid >> 5;
    const int lane = tid & 31;
    const int n0   = wave * 64;
    const int m    = lane & 15;
    const int kb   = (lane >> 4) << 3;

    v8f acc[4] = {};
    #pragma unroll
    for (int k0 = 0; k0 < DEP_; k0 += 32) {
        FragBF a;
        const float* p0 = edge_attr + (size_t)(e0 + m) * DEP_ + k0 + kb;
        const float* p1 = p0 + 16;
        #pragma unroll
        for (int i = 0; i < 8; ++i) { a.h[i] = (bf16)p0[i]; a.h[8 + i] = (bf16)p1[i]; }
        #pragma unroll
        for (int t = 0; t < 4; ++t) {
            v16bf b = load_frag_bf16(WBb, DEP_, n0 + t * 16, k0);
            acc[t] = wmma_bf16(a.v, b, acc[t]);
        }
    }
    #pragma unroll
    for (int t = 0; t < 4; ++t) {
        #pragma unroll
        for (int i = 0; i < 8; ++i) {
            int mm = i + ((lane >> 4) << 3);
            int n  = n0 + t * 16 + (lane & 15);
            float be  = acc[t][i];
            float msg = snorm[mm] * tanhf(Ax[(size_t)srow[mm] * OUT_ + n] * be);
            atomicAdd(&agg[(size_t)scol[mm] * OUT_ + n], msg);
        }
    }
}

// node = tanh( (agg + selfloop_msg) / deg + bias )
__global__ void k_node_final(const float* __restrict__ agg, const float* __restrict__ Ax,
                             const float* __restrict__ Bones, const float* __restrict__ deg,
                             const float* __restrict__ bias, float* __restrict__ node) {
    int i = blockIdx.x * blockDim.x + threadIdx.x;
    if (i >= NN_ * OUT_) return;
    int v = i >> 8, n = i & 255;
    float d    = deg[v];
    float self = tanhf(Ax[i] * Bones[n]) / d;     // norm_self = dinv^2 = 1/deg
    node[i] = tanhf((agg[i] + self) / d + bias[n]);
}

// ---------------- persistent GRU scan ------------------------------------
// One workgroup per 16-row batch tile runs all 256 timesteps on-chip.
// Per step: gh[16,768] = h[16,256] @ w_hh^T via WMMA (bf16 h mirror in LDS),
// gh exchanged wave<->wave through bf16 LDS, then gate math updates h.
__global__ __launch_bounds__(256)
void k_gru_scan(const bf16* __restrict__ Whh, const float* __restrict__ xp,
                const float* __restrict__ b_hh, float* __restrict__ seq_out,
                float* __restrict__ hT) {
    __shared__ float hs[16 * HID_];        // 16 KB  h (f32)
    __shared__ bf16  hb[16 * HID_];        //  8 KB  h (bf16 mirror for WMMA A)
    __shared__ bf16  ghs[16 * 3 * HID_];   // 24 KB  gh exchange
    const int tid  = threadIdx.x;
    const int wave = tid >> 5;
    const int lane = tid & 31;
    const int bg0  = blockIdx.x * 16;

    for (int i = tid; i < 16 * HID_; i += 256) { hs[i] = 0.f; hb[i] = (bf16)0.f; }
    __syncthreads();

    const int nW = wave * 96;              // 6 accumulation tiles per wave

    for (int t = 0; t < N_; ++t) {
        // prefetch this step's xp rows into L2/L0
        if (tid < 16)
            __builtin_prefetch(xp + ((size_t)(bg0 + tid) * N_ + t) * (3 * HID_), 0, 1);

        v8f acc[6] = {};
        for (int k0 = 0; k0 < HID_; k0 += 32) {
            v16bf a = load_frag_bf16(hb, HID_, 0, k0);
            #pragma unroll
            for (int tt = 0; tt < 6; ++tt) {
                v16bf b = load_frag_bf16(Whh, HID_, nW + tt * 16, k0);
                acc[tt] = wmma_bf16(a, b, acc[tt]);
            }
        }
        #pragma unroll
        for (int tt = 0; tt < 6; ++tt) {
            #pragma unroll
            for (int i = 0; i < 8; ++i) {
                int m = i + ((lane >> 4) << 3);
                int n = nW + tt * 16 + (lane & 15);
                ghs[m * (3 * HID_) + n] = (bf16)acc[tt][i];
            }
        }
        __syncthreads();

        // gate math: 16x256 states, 16 per thread
        #pragma unroll 4
        for (int q = 0; q < 16; ++q) {
            int id = tid + q * 256;
            int b  = id >> 8, j = id & 255;
            const float* xr = xp + ((size_t)(bg0 + b) * N_ + t) * (3 * HID_);
            float gr = (float)ghs[b * (3 * HID_) + j]            + b_hh[j];
            float gz = (float)ghs[b * (3 * HID_) + HID_ + j]     + b_hh[HID_ + j];
            float gn = (float)ghs[b * (3 * HID_) + 2 * HID_ + j] + b_hh[2 * HID_ + j];
            float r  = sigmoidf_(xr[j] + gr);
            float z  = sigmoidf_(xr[HID_ + j] + gz);
            float ng = tanhf(xr[2 * HID_ + j] + r * gn);
            float h  = (1.f - z) * ng + z * hs[b * HID_ + j];
            hs[b * HID_ + j] = h;
            hb[b * HID_ + j] = (bf16)h;
            seq_out[((size_t)(bg0 + b) * N_ + t) * HID_ + j] = h;
        }
        __syncthreads();
    }
    for (int q = 0; q < 16; ++q) {
        int id = tid + q * 256;
        int b  = id >> 8, j = id & 255;
        hT[(size_t)(bg0 + b) * HID_ + j] = hs[b * HID_ + j];
    }
}

// ---------------- pooling + classifier -----------------------------------
__global__ __launch_bounds__(256)
void k_pool(const float* __restrict__ seq, const float* __restrict__ hts,
            const float* __restrict__ lin_W, const float* __restrict__ lin_b,
            float* __restrict__ out) {
    __shared__ float pool[5 * HID_];
    __shared__ float red0[256], red1[256];
    const int b = blockIdx.x, j = threadIdx.x;

    float mx = -3.4e38f, sm = 0.f;
    for (int t = 0; t < N_; ++t) {
        float v = seq[((size_t)b * N_ + t) * HID_ + j];
        mx = fmaxf(mx, v); sm += v;
    }
    pool[j]            = hts[0 * B_ * HID_ + b * HID_ + j];
    pool[HID_ + j]     = hts[1 * B_ * HID_ + b * HID_ + j];
    pool[2 * HID_ + j] = hts[2 * B_ * HID_ + b * HID_ + j];
    pool[3 * HID_ + j] = mx;
    pool[4 * HID_ + j] = sm / (float)N_;
    __syncthreads();

    float p0 = 0.f, p1 = 0.f;
    for (int k = j; k < 5 * HID_; k += 256) {
        p0 += lin_W[k] * pool[k];
        p1 += lin_W[5 * HID_ + k] * pool[k];
    }
    red0[j] = p0; red1[j] = p1;
    __syncthreads();
    for (int s = 128; s > 0; s >>= 1) {
        if (j < s) { red0[j] += red0[j + s]; red1[j] += red1[j + s]; }
        __syncthreads();
    }
    if (j == 0) {
        float l0 = red0[0] + lin_b[0], l1 = red1[0] + lin_b[1];
        float m = fmaxf(l0, l1);
        float e0 = expf(l0 - m), e1 = expf(l1 - m), s = e0 + e1;
        out[b * 2 + 0] = e0 / s;
        out[b * 2 + 1] = e1 / s;
    }
}

// ---------------- host-side orchestration --------------------------------
extern "C" void kernel_launch(void* const* d_in, const int* in_sizes, int n_in,
                              void* d_out, int out_size, void* d_ws, size_t ws_size,
                              hipStream_t stream) {
    (void)in_sizes; (void)n_in; (void)out_size; (void)ws_size;

    const float* x         = (const float*)d_in[0];
    const float* edge_attr = (const float*)d_in[1];
    const int*   edge_index= (const int*)  d_in[2];
    const float* WA        = (const float*)d_in[3];
    const float* WB        = (const float*)d_in[4];
    const float* gcn_bias  = (const float*)d_in[5];
    const float* w_ih[3]   = {(const float*)d_in[6],  (const float*)d_in[10], (const float*)d_in[14]};
    const float* w_hh[3]   = {(const float*)d_in[7],  (const float*)d_in[11], (const float*)d_in[15]};
    const float* b_ih[3]   = {(const float*)d_in[8],  (const float*)d_in[12], (const float*)d_in[16]};
    const float* b_hh[3]   = {(const float*)d_in[9],  (const float*)d_in[13], (const float*)d_in[17]};
    const float* lin_W     = (const float*)d_in[18];
    const float* lin_b     = (const float*)d_in[19];
    float* out = (float*)d_out;

    // workspace carving (256B aligned)
    char* ws = (char*)d_ws;
    size_t off = 0;
    auto take = [&](size_t bytes) { size_t o = off; off += (bytes + 255) & ~(size_t)255; return o; };

    float* deg   = (float*)(ws + take((size_t)NN_ * 4));
    bf16*  xb    = (bf16*) (ws + take((size_t)NN_ * FIN_ * 2));
    bf16*  WAb   = (bf16*) (ws + take((size_t)OUT_ * FIN_ * 2));
    bf16*  WBb   = (bf16*) (ws + take((size_t)OUT_ * DEP_ * 2));
    bf16*  wihb[3]; bf16* whhb[3];
    for (int l = 0; l < 3; ++l) {
        wihb[l] = (bf16*)(ws + take((size_t)3 * HID_ * HID_ * 2));
        whhb[l] = (bf16*)(ws + take((size_t)3 * HID_ * HID_ * 2));
    }
    float* Ax    = (float*)(ws + take((size_t)NN_ * OUT_ * 4));
    float* agg   = (float*)(ws + take((size_t)NN_ * OUT_ * 4));
    float* Bones = (float*)(ws + take((size_t)OUT_ * 4));
    float* bufA  = (float*)(ws + take((size_t)NN_ * HID_ * 4));
    float* bufB  = (float*)(ws + take((size_t)NN_ * HID_ * 4));
    bf16*  seqb  = (bf16*) (ws + take((size_t)NN_ * HID_ * 2));
    float* xp    = (float*)(ws + take((size_t)NN_ * 3 * HID_ * 4));
    float* hts   = (float*)(ws + take((size_t)3 * B_ * HID_ * 4));

    // ---- GCN ----
    k_fill_f32 <<<(NN_ + 255) / 256, 256, 0, stream>>>(deg, 1.0f, NN_);        // self loops
    k_edge_deg <<<(E_ + 255) / 256, 256, 0, stream>>>(edge_index, deg);
    k_fill_f32 <<<((NN_ * OUT_) + 255) / 256, 256, 0, stream>>>(agg, 0.f, NN_ * OUT_);

    k_convert_bf16<<<(NN_ * FIN_ + 255) / 256, 256, 0, stream>>>(x,  xb,  NN_ * FIN_);
    k_convert_bf16<<<(OUT_ * FIN_ + 255) / 256, 256, 0, stream>>>(WA, WAb, OUT_ * FIN_);
    k_convert_bf16<<<(OUT_ * DEP_ + 255) / 256, 256, 0, stream>>>(WB, WBb, OUT_ * DEP_);
    for (int l = 0; l < 3; ++l) {
        k_convert_bf16<<<(3 * HID_ * HID_ + 255) / 256, 256, 0, stream>>>(w_ih[l], wihb[l], 3 * HID_ * HID_);
        k_convert_bf16<<<(3 * HID_ * HID_ + 255) / 256, 256, 0, stream>>>(w_hh[l], whhb[l], 3 * HID_ * HID_);
    }
    k_bones<<<1, 256, 0, stream>>>(WB, Bones);

    dim3 gAx(NN_ / 16, OUT_ / 256);                         // (1024,1)
    k_gemm_bf16<<<gAx, 128, 0, stream>>>(xb, WAb, nullptr, Ax, NN_, OUT_, FIN_);

    k_edge_msg <<<E_ / 16, 128, 0, stream>>>(edge_attr, edge_index, deg, Ax, WBb, agg);
    k_node_final<<<(NN_ * OUT_) / 256, 256, 0, stream>>>(agg, Ax, Bones, deg, gcn_bias, bufA);

    // ---- 3-layer GRU ----
    float* cur = bufA;
    float* nxt = bufB;
    for (int l = 0; l < 3; ++l) {
        k_convert_bf16<<<(NN_ * HID_ + 255) / 256, 256, 0, stream>>>(cur, seqb, NN_ * HID_);
        dim3 gxp(NN_ / 16, (3 * HID_) / 256);               // (1024,3)
        k_gemm_bf16<<<gxp, 128, 0, stream>>>(seqb, wihb[l], b_ih[l], xp, NN_, 3 * HID_, HID_);
        k_gru_scan<<<B_ / 16, 256, 0, stream>>>(whhb[l], xp, b_hh[l], nxt, hts + (size_t)l * B_ * HID_);
        float* tmp = cur; cur = nxt; nxt = tmp;
    }

    // ---- pooling + classifier ----
    k_pool<<<B_, 256, 0, stream>>>(cur, hts, lin_W, lin_b, out);
}
